// TrajGRU_45174466019752
// MI455X (gfx1250) — compile-verified, matrix operand order checked by reading, not run on hardware
//
#include <hip/hip_runtime.h>
#include <hip/hip_bf16.h>
#include <math.h>

typedef __attribute__((ext_vector_type(16))) _Float16 v16h;
typedef __attribute__((ext_vector_type(8)))  _Float16 v8h;
typedef __attribute__((ext_vector_type(8)))  float    v8f;

#define HW_   16384
#define W_    128
#define H_    128
#define BB_   4
#define CH_   64
#define LL_   5

static __host__ __device__ inline int ceil16(int v) { return (v + 15) & ~15; }
static __host__ __device__ inline int ceil32(int v) { return (v + 31) & ~31; }

// ---------------------------------------------------------------------------
// Implicit-GEMM conv via V_WMMA_F32_16X16X32_F16, register-blocked over N.
// One wave computes 16 pixels x (NT*16) output channels of one image.
//
// Weights are pre-staged in f16 with zero padding:
//   rows 0..Npad-1 (Npad = ceil16(cout)), row stride Kpad = ceil32(cin*KSZ*KSZ),
//   zero outside [cout, Ktot). Consequences:
//   - B fragment = two unconditional 16-byte loads per lane (global_load_b128);
//   - A needs NO k-bounds select: K-tail products hit zero B columns;
//   - only the spatial border select remains (KSZ==3), done branchlessly.
//
// Fragment layouts per CDNA5 ISA (wave32):
//   A/B: lane = (half = lane>>4, nl = lane&15); element e <-> k-offset
//        (e&7) + 8*half + 16*(e>>3)  == two runs of 8: [8h..8h+7],[16+8h..+7]
//   C/D: n = lane&15; VGPR r <-> m = r + 8*half
// ---------------------------------------------------------------------------
template <int KSZ, int PAD, int NT>
__global__ void wmma_conv_kernel(const float* __restrict__ in, int cin, long in_bstride,
                                 const _Float16* __restrict__ w, const float* __restrict__ bias,
                                 int cout, float* __restrict__ out, long out_bstride)
{
    const int lane = threadIdx.x;
    const int half = lane >> 4;
    const int nl   = lane & 15;

    const int tileM = blockIdx.x;        // 1024 strips of 16 pixels along W
    const int tileN = blockIdx.y;
    const int b     = blockIdx.z;

    constexpr int KK = KSZ * KSZ;
    const int Ktot = cin * KK;
    const int Kpad = ceil32(Ktot);

    const float* inb = in + (long)b * in_bstride;

    const int p0 = tileM * 16;
    const int y  = p0 / W_;              // W_ % 16 == 0: strip stays in one row
    const int x0 = p0 % W_;
    const int px = x0 + nl;              // this lane's A-row pixel x

    int co[NT];
#pragma unroll
    for (int nt = 0; nt < NT; ++nt)
        co[nt] = (tileN * NT + nt) * 16 + nl;   // B-row / D-column channel

    v8f acc[NT];
#pragma unroll
    for (int nt = 0; nt < NT; ++nt) acc[nt] = (v8f){};

    for (int kb = 0; kb < Kpad; kb += 32) {
        // Prefetch next weight chunk (speculative; lowers to global_prefetch_b8).
        __builtin_prefetch(w + (long)co[0] * Kpad + kb + 32, 0, 1);

        // ---- A fragment: 16 im2col patch values, all loads unconditional ----
        v16h a;
#pragma unroll
        for (int e = 0; e < 16; ++e) {
            const int k  = kb + (e & 7) + 8 * half + 16 * (e >> 3);
            const int kc = min(k, Ktot - 1);           // safe address for K-tail
            float av;
            if constexpr (KSZ == 1) {
                av = inb[(long)kc * HW_ + p0 + nl];
            } else {
                const int ci = kc / KK;                // constant divisors
                const int r  = kc % KK;
                const int kh = r / KSZ;
                const int kw = r % KSZ;
                const int iy = y  + kh - PAD;
                const int ix = px + kw - PAD;
                const bool valid = ((unsigned)iy < (unsigned)H_) &
                                   ((unsigned)ix < (unsigned)W_);
                const float v = inb[(long)ci * HW_ + (iy & (H_ - 1)) * W_ + (ix & (W_ - 1))];
                av = valid ? v : 0.0f;                 // branchless select
            }
            a[e] = (_Float16)av;
        }

        // ---- NT B fragments (2 x b128 each) + WMMAs, reusing A ----
#pragma unroll
        for (int nt = 0; nt < NT; ++nt) {
            const _Float16* wrow = w + (long)co[nt] * Kpad + kb + 8 * half;
            const v8h b0 = *(const v8h*)(wrow);
            const v8h b1 = *(const v8h*)(wrow + 16);
            v16h bm;
#pragma unroll
            for (int e = 0; e < 8; ++e) { bm[e] = b0[e]; bm[8 + e] = b1[e]; }
            acc[nt] = __builtin_amdgcn_wmma_f32_16x16x32_f16(
                false, a, false, bm, (short)0, acc[nt], false, false);
        }
    }

#pragma unroll
    for (int nt = 0; nt < NT; ++nt) {
        if (co[nt] < cout) {
            const float bb = bias ? bias[co[nt]] : 0.0f;
            float* outb = out + (long)b * out_bstride + (long)co[nt] * HW_;
#pragma unroll
            for (int r = 0; r < 8; ++r) {
                const int pp = p0 + r + 8 * half;
                outb[pp] = acc[nt][r] + bb;
            }
        }
    }
}

// ---------------------------------------------------------------------------
// f32 -> f16 weight staging with zero padding to (Npad x Kpad).
// ---------------------------------------------------------------------------
__global__ void pad_cvt_f16_kernel(const float* __restrict__ src, _Float16* __restrict__ dst,
                                   int cout, int Ktot, int Kpad, long total)
{
    long idx = (long)blockIdx.x * blockDim.x + threadIdx.x;
    if (idx >= total) return;
    const int co = (int)(idx / Kpad);
    const int k  = (int)(idx % Kpad);
    float v = 0.0f;
    if (co < cout && k < Ktot) v = src[(long)co * Ktot + k];
    dst[idx] = (_Float16)v;
}

// ---------------------------------------------------------------------------
// concat([src1(B,c1,H,W with batch stride bs1), src2(B,c2,H,W)]) -> dst
// ---------------------------------------------------------------------------
__global__ void concat_kernel(const float* __restrict__ s1, int c1, long bs1,
                              const float* __restrict__ s2, int c2,
                              float* __restrict__ dst)
{
    const int ct = c1 + c2;
    const long total = (long)BB_ * ct * HW_;
    long idx = (long)blockIdx.x * blockDim.x + threadIdx.x;
    if (idx >= total) return;
    const int p = (int)(idx % HW_);
    const int t = (int)(idx / HW_);
    const int c = t % ct;
    const int b = t / ct;
    float v;
    if (c < c1) v = s1[(long)b * bs1 + (long)c * HW_ + p];
    else        v = s2[((long)b * c2 + (c - c1)) * HW_ + p];
    dst[idx] = v;
}

// ---------------------------------------------------------------------------
// Bilinear warp (reference semantics: clamp sample point, border padding).
// One thread per (b, l, pixel); loops over 64 channels reusing indices/weights.
// ---------------------------------------------------------------------------
__global__ void warp_kernel(const float* __restrict__ h, const float* __restrict__ flow,
                            float* __restrict__ hcat)
{
    const long total = (long)BB_ * LL_ * HW_;
    long idx = (long)blockIdx.x * blockDim.x + threadIdx.x;
    if (idx >= total) return;
    const int p = (int)(idx % HW_);
    const int t = (int)(idx / HW_);
    const int l = t % LL_;
    const int b = t / LL_;
    const int yy = p / W_;
    const int xx = p % W_;

    const float* fb = flow + ((long)b * (2 * LL_) + 2 * l) * HW_;
    const float dx = fb[p];
    const float dy = fb[HW_ + p];

    float px = fminf(fmaxf((float)xx + dx, 0.0f), (float)(W_ - 1));
    float py = fminf(fmaxf((float)yy + dy, 0.0f), (float)(H_ - 1));
    const float x0f = floorf(px);
    const float y0f = floorf(py);
    const float wx = px - x0f;
    const float wy = py - y0f;
    const int x0 = (int)x0f;
    const int y0 = (int)y0f;
    const int x1 = min(x0 + 1, W_ - 1);
    const int y1 = min(y0 + 1, H_ - 1);
    const int i00 = y0 * W_ + x0;
    const int i01 = y0 * W_ + x1;
    const int i10 = y1 * W_ + x0;
    const int i11 = y1 * W_ + x1;
    const float w00 = (1.0f - wx) * (1.0f - wy);
    const float w01 = wx * (1.0f - wy);
    const float w10 = (1.0f - wx) * wy;
    const float w11 = wx * wy;

    const float* hb = h + (long)b * CH_ * HW_;
    float* ob = hcat + ((long)b * LL_ * CH_ + (long)l * CH_) * HW_;
    for (int c = 0; c < CH_; ++c) {
        const float* hc = hb + (long)c * HW_;
        ob[(long)c * HW_ + p] = w00 * hc[i00] + w01 * hc[i01]
                              + w10 * hc[i10] + w11 * hc[i11];
    }
}

// ---------------------------------------------------------------------------
// Gates: i = sigmoid(xc0+hc0), f = sigmoid(xc1+hc1), o = tanh(xc2+hc2)
//        h_new = f * h_agg + i * o
// ---------------------------------------------------------------------------
__global__ void gates_kernel(const float* __restrict__ xc, const float* __restrict__ hc,
                             const float* __restrict__ hagg, float* __restrict__ hout)
{
    const long total = (long)BB_ * CH_ * HW_;
    long idx = (long)blockIdx.x * blockDim.x + threadIdx.x;
    if (idx >= total) return;
    const int p = (int)(idx % HW_);
    const int t = (int)(idx / HW_);
    const int c = t % CH_;
    const int b = t / CH_;

    const long base3 = (long)b * (3 * CH_) * HW_;
    const long o0 = base3 + (long)c * HW_ + p;
    const long o1 = base3 + (long)(CH_ + c) * HW_ + p;
    const long o2 = base3 + (long)(2 * CH_ + c) * HW_ + p;

    const float iv = xc[o0] + hc[o0];
    const float fv = xc[o1] + hc[o1];
    const float ov = xc[o2] + hc[o2];

    const float ig = 1.0f / (1.0f + __expf(-iv));
    const float fg = 1.0f / (1.0f + __expf(-fv));
    const float og = tanhf(ov);

    const long hb = ((long)b * CH_ + c) * HW_ + p;
    hout[hb] = fg * hagg[hb] + ig * og;
}

// ---------------------------------------------------------------------------
// Host orchestration
// ---------------------------------------------------------------------------
struct CellP {
    const float *x_w, *x_b, *h_w, *h_b, *flow_w, *flow_b, *warp_w, *warp_b; // f32 inputs
    const _Float16 *x_wh, *h_wh, *flow_wh, *warp_wh;                        // padded f16
};

static _Float16* stage_w(const float* src, _Float16* dst, int cout, int Ktot,
                         hipStream_t stream)
{
    const int Npad = ceil16(cout);
    const int Kpad = ceil32(Ktot);
    const long total = (long)Npad * Kpad;
    pad_cvt_f16_kernel<<<dim3((unsigned)((total + 255) / 256)), 256, 0, stream>>>(
        src, dst, cout, Ktot, Kpad, total);
    return dst + total;
}

static void cell_step(const float* xin, int cin, long xbstride, const CellP& P,
                      float* hbuf, float* CAT, float* FLOW, float* HCAT,
                      float* HAGG, float* XC, float* HC, hipStream_t stream)
{
    const int ctot = cin + CH_;
    // 1) concat(x, h)
    {
        long n = (long)BB_ * ctot * HW_;
        concat_kernel<<<dim3((unsigned)((n + 255) / 256)), 256, 0, stream>>>(
            xin, cin, xbstride, hbuf, CH_, CAT);
    }
    // 2) flows = conv3x3(CAT) -> 10 channels           (KSZ=3, NT=1)
    wmma_conv_kernel<3, 1, 1><<<dim3(HW_ / 16, 1, BB_), 32, 0, stream>>>(
        CAT, ctot, (long)ctot * HW_, P.flow_wh, P.flow_b,
        2 * LL_, FLOW, (long)2 * LL_ * HW_);
    // 3) warp h by L flows -> HCAT (B, 320, H, W)
    {
        long n = (long)BB_ * LL_ * HW_;
        warp_kernel<<<dim3((unsigned)((n + 255) / 256)), 256, 0, stream>>>(hbuf, FLOW, HCAT);
    }
    // 4) h_agg = conv1x1(HCAT, K=320) -> 64            (KSZ=1, NT=4, one N block)
    wmma_conv_kernel<1, 0, 4><<<dim3(HW_ / 16, 1, BB_), 32, 0, stream>>>(
        HCAT, LL_ * CH_, (long)LL_ * CH_ * HW_, P.warp_wh, P.warp_b,
        CH_, HAGG, (long)CH_ * HW_);
    // 5) xc = conv3x3(x) -> 192                        (KSZ=3, NT=4, 3 N blocks)
    wmma_conv_kernel<3, 1, 4><<<dim3(HW_ / 16, 3, BB_), 32, 0, stream>>>(
        xin, cin, xbstride, P.x_wh, P.x_b,
        3 * CH_, XC, (long)3 * CH_ * HW_);
    // 6) hc = conv3x3(h_agg) -> 192
    wmma_conv_kernel<3, 1, 4><<<dim3(HW_ / 16, 3, BB_), 32, 0, stream>>>(
        HAGG, CH_, (long)CH_ * HW_, P.h_wh, P.h_b,
        3 * CH_, HC, (long)3 * CH_ * HW_);
    // 7) gates -> h (in place)
    {
        long n = (long)BB_ * CH_ * HW_;
        gates_kernel<<<dim3((unsigned)((n + 255) / 256)), 256, 0, stream>>>(XC, HC, HAGG, hbuf);
    }
}

extern "C" void kernel_launch(void* const* d_in, const int* in_sizes, int n_in,
                              void* d_out, int out_size, void* d_ws, size_t ws_size,
                              hipStream_t stream)
{
    (void)out_size; (void)ws_size;
    if (n_in < 19) return;

    const int Tin = 10, Tout = 4;
    const float* x = nullptr;
    const float *out_w = nullptr, *out_b = nullptr;
    CellP c0 = {}, c1 = {};

    if (in_sizes[0] > 1000) {
        // Insertion order: x, cell0{x_w,x_b,h_w,h_b,flow_w,flow_b,warp_w,warp_b},
        //                  cell1{...}, out_w, out_b
        x = (const float*)d_in[0];
        c0.x_w    = (const float*)d_in[1];  c0.x_b    = (const float*)d_in[2];
        c0.h_w    = (const float*)d_in[3];  c0.h_b    = (const float*)d_in[4];
        c0.flow_w = (const float*)d_in[5];  c0.flow_b = (const float*)d_in[6];
        c0.warp_w = (const float*)d_in[7];  c0.warp_b = (const float*)d_in[8];
        c1.x_w    = (const float*)d_in[9];  c1.x_b    = (const float*)d_in[10];
        c1.h_w    = (const float*)d_in[11]; c1.h_b    = (const float*)d_in[12];
        c1.flow_w = (const float*)d_in[13]; c1.flow_b = (const float*)d_in[14];
        c1.warp_w = (const float*)d_in[15]; c1.warp_b = (const float*)d_in[16];
        out_w = (const float*)d_in[17];
        out_b = (const float*)d_in[18];
    } else {
        // Sorted pytree leaves: cells (sorted keys: flow_b, flow_w, h_b, h_w,
        // warp_b, warp_w, x_b, x_w), then out_b, out_w, x.
        c0.flow_b = (const float*)d_in[0];  c0.flow_w = (const float*)d_in[1];
        c0.h_b    = (const float*)d_in[2];  c0.h_w    = (const float*)d_in[3];
        c0.warp_b = (const float*)d_in[4];  c0.warp_w = (const float*)d_in[5];
        c0.x_b    = (const float*)d_in[6];  c0.x_w    = (const float*)d_in[7];
        c1.flow_b = (const float*)d_in[8];  c1.flow_w = (const float*)d_in[9];
        c1.h_b    = (const float*)d_in[10]; c1.h_w    = (const float*)d_in[11];
        c1.warp_b = (const float*)d_in[12]; c1.warp_w = (const float*)d_in[13];
        c1.x_b    = (const float*)d_in[14]; c1.x_w    = (const float*)d_in[15];
        out_b = (const float*)d_in[16];
        out_w = (const float*)d_in[17];
        x     = (const float*)d_in[18];
    }

    // Workspace layout (floats). HCAT overlaps XC/HC region (disjoint lifetimes).
    float* ws   = (float*)d_ws;
    float* h0   = ws;                        //  4*64*HW   = 4,194,304
    float* h1   = ws + 4194304;              //  4,194,304
    float* CAT  = ws + 8388608;              //  4*128*HW  = 8,388,608
    float* FLOW = ws + 16777216;             //  4*10*HW   =   655,360
    float* HAGG = ws + 17432576;             //  4*64*HW   = 4,194,304
    float* XC   = ws + 21626880;             //  4*192*HW  = 12,582,912
    float* HC   = XC + 12582912;             //  4*192*HW  = 12,582,912
    float* HCAT = XC;                        //  4*320*HW fits in XC+HC (25.1M floats)
    _Float16* wh = (_Float16*)(ws + 46792704);   // 16-byte aligned

    // Zero hidden states (h0 and h1 contiguous; capture-safe async memset).
    hipMemsetAsync(h0, 0, (size_t)2 * 4194304 * sizeof(float), stream);

    // One-time (per launch) padded f16 weight staging.
    _Float16* p = wh;
    c0.x_wh    = p; p = stage_w(c0.x_w,    p, 3 * CH_,  1 * 9,            stream);
    c0.h_wh    = p; p = stage_w(c0.h_w,    p, 3 * CH_,  CH_ * 9,          stream);
    c0.flow_wh = p; p = stage_w(c0.flow_w, p, 2 * LL_,  (1 + CH_) * 9,    stream);
    c0.warp_wh = p; p = stage_w(c0.warp_w, p, CH_,      LL_ * CH_,        stream);
    c1.x_wh    = p; p = stage_w(c1.x_w,    p, 3 * CH_,  CH_ * 9,          stream);
    c1.h_wh    = p; p = stage_w(c1.h_w,    p, 3 * CH_,  CH_ * 9,          stream);
    c1.flow_wh = p; p = stage_w(c1.flow_w, p, 2 * LL_,  (CH_ + CH_) * 9,  stream);
    c1.warp_wh = p; p = stage_w(c1.warp_w, p, CH_,      LL_ * CH_,        stream);
    const _Float16* out_wh = p; p = stage_w(out_w, p, 1, CH_, stream);

    // Warm-up over input sequence.
    for (int t = 0; t < Tin; ++t) {
        const float* xt = x + (long)t * HW_;           // x: (B,1,Tin,H,W)
        cell_step(xt, 1, (long)Tin * HW_, c0, h0, CAT, FLOW, HCAT, HAGG, XC, HC, stream);
        cell_step(h0, CH_, (long)CH_ * HW_, c1, h1, CAT, FLOW, HCAT, HAGG, XC, HC, stream);
    }

    // Autoregressive prediction.
    float* dout = (float*)d_out;                       // (B,1,Tout,H,W)
    for (int t = 0; t < Tout; ++t) {
        const float* pin;
        long ps;
        if (t == 0) { pin = x + (long)(Tin - 1) * HW_; ps = (long)Tin * HW_; }
        else        { pin = dout + (long)(t - 1) * HW_; ps = (long)Tout * HW_; }
        cell_step(pin, 1, ps, c0, h0, CAT, FLOW, HCAT, HAGG, XC, HC, stream);
        cell_step(h0, CH_, (long)CH_ * HW_, c1, h1, CAT, FLOW, HCAT, HAGG, XC, HC, stream);
        // out conv 1x1: 64 -> 1, written straight into d_out slice t.   (KSZ=1, NT=1)
        wmma_conv_kernel<1, 0, 1><<<dim3(HW_ / 16, 1, BB_), 32, 0, stream>>>(
            h1, CH_, (long)CH_ * HW_, out_wh, out_b,
            1, dout + (long)t * HW_, (long)Tout * HW_);
    }
}
